// StaticTraceRobotApp_73040213836040
// MI455X (gfx1250) — matrix operand order repaired and checked by reading, not compiled, not run on hardware
//
#include <hip/hip_runtime.h>

typedef __attribute__((ext_vector_type(16))) _Float16 v16h;
typedef __attribute__((ext_vector_type(8)))  float    v8f;
typedef __attribute__((ext_vector_type(4)))  float    v4f;

// CDNA5 16-bit A-matrix (16x32) K index for half-slot h, lane-group g (ISA 7.12.2):
// h<8 -> K = h + 8g ; h>=8 -> K = (h-8) + 16 + 8g  => two contiguous 8-float runs.

// Branchless predicated loads: unconditional load from a clamped index, then
// select. Avoids per-element saveexec/branch/waitcnt serialization.
__device__ __forceinline__ float gload_or_zero(const float* __restrict__ p,
                                               int idx, bool valid) {
    const int i = valid ? idx : 0;
    const float v = p[i];
    return valid ? v : 0.0f;
}
__device__ __forceinline__ v4f gload4_or_zero(const float* __restrict__ p,
                                              int idx4, bool valid) {
    const v4f* __restrict__ q = (const v4f*)p;
    const int i = valid ? idx4 : 0;
    const v4f v = q[i];
    const v4f z = {0.0f, 0.0f, 0.0f, 0.0f};
    return valid ? v : z;
}

__global__ __launch_bounds__(256) void robot_fused_kernel(
    const float* __restrict__ x,       // [3,64,64] sensor
    const float* __restrict__ pose0,   // [3]
    const float* __restrict__ ang0,    // [2]
    const float* __restrict__ ticks,   // [2]
    const float* __restrict__ conv_w,  // [4,3,4,4] -> flat [oc*48 + ic*16 + ky*4 + kx]
    const float* __restrict__ conv_b,  // [4]
    const float* __restrict__ lin_w,   // [2,256]
    const float* __restrict__ lin_b,   // [2]
    float* __restrict__ out)           // [41] = trajectory[19*2] ++ pose[3]
{
    __shared__ float conv_lds[1024];   // [4][16][16] relu(conv)
    __shared__ float pooled[256];      // channel-major flatten of [4][8][8]
    __shared__ float tgt[2];

    const int tid  = threadIdx.x;
    const int lane = tid & 31;
    const int wave = tid >> 5;
    const int n    = lane & 15;        // N (B/D column) or M (A row)
    const int g    = lane >> 4;        // lane half-group

    // Pull the 48KB sensor image toward the WGP (global_prefetch_b8).
    __builtin_prefetch(x + tid * 48, 0, 3);

    // ---------------- Phase 1: conv as implicit-im2col WMMA GEMM ----------------
    // D[16x16] += A[16x32] * B[32x16]; M=oc (4 used), K=ic*16+ky*4+kx (48 used, two
    // K=32 chunks), N=16 output positions per tile; 16 tiles over 8 waves.

    // A (weights): per lane three valid 8-float runs -> six aligned b128 loads.
    // Run bases (floats): n*48 + {8g, 16+8g, 32+8g}; run K>=48 is all zero.
    const bool wa = (n < 4);
    const int wb4 = n * 12 + 2 * g;                 // float4 index of run 0
    const v4f w00 = gload4_or_zero(conv_w, wb4,     wa);
    const v4f w01 = gload4_or_zero(conv_w, wb4 + 1, wa);
    const v4f w10 = gload4_or_zero(conv_w, wb4 + 4, wa);
    const v4f w11 = gload4_or_zero(conv_w, wb4 + 5, wa);
    const v4f w20 = gload4_or_zero(conv_w, wb4 + 8, wa);
    const v4f w21 = gload4_or_zero(conv_w, wb4 + 9, wa);
    v16h a0 = {}, a1 = {};
    #pragma unroll
    for (int h = 0; h < 16; ++h) {
        const float lo = (h < 8) ? ((h < 4)  ? w00[h & 3] : w01[h & 3])
                                 : ((h < 12) ? w10[h & 3] : w11[h & 3]);
        a0[h] = (_Float16)lo;
        a1[h] = (_Float16)((h < 8) ? ((h < 4) ? w20[h & 3] : w21[h & 3]) : 0.0f);
    }

    #pragma unroll
    for (int t = 0; t < 2; ++t) {
        const int p  = (wave * 2 + t) * 16 + n;   // output position oy*16+ox
        const int oy = p >> 4, ox = p & 15;
        // B slot h holds K = chunk*32 + kk, kk = 8*g + h + (h&8) (table order); with
        // K = ic*16 + ky*4 + kx: chunk0 -> ic=kk>>4, chunk1 -> ic=2+(kk>>4) (kk<16 valid).
        // Load phase (plain floats) kept separate from convert phase so the
        // scheduler can clause all gathers under one wait.
        float bf0[16], bf1[16];
        #pragma unroll
        for (int h = 0; h < 16; ++h) {
            const int kk = 8 * g + h + (h & 8), ky = (kk >> 2) & 3, kx = kk & 3;
            const int iy = oy * 4 - 1 + ky;       // max 62 -> only lower bound matters
            const int ix = ox * 4 - 1 + kx;
            const bool inb = (iy >= 0) && (ix >= 0);
            const int  ofs = iy * 64 + ix;
            bf0[h] = gload_or_zero(x, (kk >> 4) * 4096 + ofs,  inb);
            bf1[h] = gload_or_zero(x, 2 * 4096 + ofs, inb && (kk < 16));
        }
        v16h b0 = {}, b1 = {};
        #pragma unroll
        for (int h = 0; h < 16; ++h) {
            b0[h] = (_Float16)bf0[h];
            b1[h] = (_Float16)bf1[h];
        }
        v8f acc = {};
        acc = __builtin_amdgcn_wmma_f32_16x16x32_f16(false, a0, false, b0,
                                                     (short)0, acc, false, false);
        acc = __builtin_amdgcn_wmma_f32_16x16x32_f16(false, a1, false, b1,
                                                     (short)0, acc, false, false);
        // D: lanes 0-15 hold M=0..7 in c[0..7]; channels 0..3 live in c[0..3].
        if (g == 0) {
            #pragma unroll
            for (int oc = 0; oc < 4; ++oc)
                conv_lds[oc * 256 + p] = fmaxf(acc[oc] + conv_b[oc], 0.0f);
        }
    }
    __syncthreads();

    // ---------------- Phase 2: 2x2 maxpool + channel-major flatten ----------------
    {
        const int c = tid >> 6, r = tid & 63, py = r >> 3, px = r & 7;
        const float* base = conv_lds + c * 256 + (2 * py) * 16 + 2 * px;
        pooled[tid] = fmaxf(fmaxf(base[0], base[1]), fmaxf(base[16], base[17]));
    }
    __syncthreads();

    // ---------------- Phase 3: FC 256->2 on wave 0 via chained WMMA ----------------
    if (wave == 0) {
        const bool la = (n < 2);
        v8f acc = {};
        #pragma unroll
        for (int k0 = 0; k0 < 256; k0 += 32) {
            // A runs (floats): n*256 + k0 + {8g, 16+8g} -> four aligned b128 loads.
            const int lb4 = n * 64 + (k0 >> 2) + 2 * g;
            const v4f l0 = gload4_or_zero(lin_w, lb4,     la);
            const v4f l1 = gload4_or_zero(lin_w, lb4 + 1, la);
            const v4f l2 = gload4_or_zero(lin_w, lb4 + 4, la);
            const v4f l3 = gload4_or_zero(lin_w, lb4 + 5, la);
            v16h a = {}, b = {};
            #pragma unroll
            for (int h = 0; h < 16; ++h) {
                const float av = (h < 8) ? ((h < 4)  ? l0[h & 3] : l1[h & 3])
                                         : ((h < 12) ? l2[h & 3] : l3[h & 3]);
                a[h] = (_Float16)av;
                const float bv = pooled[k0 + 8 * g + h + (h & 8)];   // always in range
                b[h] = (_Float16)((n == 0) ? bv : 0.0f);
            }
            acc = __builtin_amdgcn_wmma_f32_16x16x32_f16(false, a, false, b,
                                                         (short)0, acc, false, false);
        }
        if (lane == 0) {            // D[m][0] in c[m] of lane 0
            tgt[0] = acc[0] + lin_b[0];
            tgt[1] = acc[1] + lin_b[1];
        }
    }
    __syncthreads();

    // ---------------- Phase 4: serial tail (odometry, IK x9, trajectory) ----------
    if (tid == 0) {
        // Odometry
        const float dl = ticks[0] * 1e-4f, dr = ticks[1] * 1e-4f;
        const float dc = 0.5f * (dl + dr);
        const float dth = (dr - dl) / 0.5f;
        const float avg = pose0[2] + 0.5f * dth;
        const float npx = pose0[0] + dc * cosf(avg);
        const float npy = pose0[1] + dc * sinf(avg);
        const float nth = pose0[2] + dth;

        // IK (L1 = L2 = 1, ALPHA = 1)
        float q1 = ang0[0], q2 = ang0[1];
        const float tx = tgt[0], ty = tgt[1];
        for (int i = 0; i < 9; ++i) {
            const float s1 = sinf(q1), c1 = cosf(q1);
            const float s12 = sinf(q1 + q2), c12 = cosf(q1 + q2);
            const float ex = tx - (c1 + c12);
            const float ey = ty - (s1 + s12);
            const float j11 = -s1 - s12, j12 = -s12;
            const float j21 = c1 + c12,  j22 = c12;
            const float inv = 1.0f / (j11 * j22 - j12 * j21 + 1e-6f);
            q1 += inv * ( j22 * ex - j12 * ey);
            q2 += inv * (-j21 * ex + j11 * ey);
        }

        // Trajectory: start -> start+(-0.2,0) -> +(0,-0.1), 10+9 points
        const float sx = cosf(q1) + cosf(q1 + q2);
        const float sy = sinf(q1) + sinf(q1 + q2);
        const float mx = sx - 0.2f, my = sy;
        const float ex2 = mx,       ey2 = my - 0.1f;
        for (int i = 0; i < 10; ++i) {
            const float t = (float)i / 9.0f;
            out[2 * i + 0] = sx * (1.0f - t) + mx * t;
            out[2 * i + 1] = sy * (1.0f - t) + my * t;
        }
        for (int i = 1; i < 10; ++i) {
            const float t = (float)i / 9.0f;
            out[2 * (9 + i) + 0] = mx * (1.0f - t) + ex2 * t;
            out[2 * (9 + i) + 1] = my * (1.0f - t) + ey2 * t;
        }
        out[38] = npx; out[39] = npy; out[40] = nth;
    }
}

extern "C" void kernel_launch(void* const* d_in, const int* in_sizes, int n_in,
                              void* d_out, int out_size, void* d_ws, size_t ws_size,
                              hipStream_t stream) {
    (void)in_sizes; (void)n_in; (void)d_ws; (void)ws_size; (void)out_size;
    const float* x      = (const float*)d_in[0];
    const float* pose0  = (const float*)d_in[1];
    const float* ang0   = (const float*)d_in[2];
    const float* ticks  = (const float*)d_in[3];
    const float* conv_w = (const float*)d_in[4];
    const float* conv_b = (const float*)d_in[5];
    const float* lin_w  = (const float*)d_in[6];
    const float* lin_b  = (const float*)d_in[7];
    float* out = (float*)d_out;

    // Entire pipeline is latency-bound (~50KB traffic, ~2e5 flops): one launch,
    // one workgroup (8 wave32s on a single WGP), fused through LDS.
    robot_fused_kernel<<<1, 256, 0, stream>>>(x, pose0, ang0, ticks,
                                              conv_w, conv_b, lin_w, lin_b, out);
}
